// MistralAttention_64819646431913
// MI455X (gfx1250) — matrix operand order repaired and checked by reading, not compiled, not run on hardware
//
#include <hip/hip_runtime.h>

typedef __attribute__((ext_vector_type(16))) __bf16 v16bf;
typedef __attribute__((ext_vector_type(8)))  __bf16 v8bf;
typedef __attribute__((ext_vector_type(8)))  float  v8f;

#define S_    2048
#define H_    4096
#define NH_   32
#define NKV_  8
#define D_    128
#define WIN_  1024
#define QKVW  ((NH_ + 2 * NKV_) * D_)   // 6144

// ---------------------------------------------------------------------------
// WMMA operand loaders (wave32, v_wmma_f32_16x16x32_bf16 layouts per ISA 7.12.2)
// ---------------------------------------------------------------------------
__device__ __forceinline__ v16bf cat8(v8bf a, v8bf b) {
  return __builtin_shufflevector(a, b, 0,1,2,3,4,5,6,7,8,9,10,11,12,13,14,15);
}

// A (16xK slice of row-major matrix): lane m=lane&15 holds row m.
// hi = lane>>4 selects K groups {hi*8..hi*8+7} and {16+hi*8..16+hi*8+7}.
__device__ __forceinline__ v16bf load_tileA(const __bf16* base, int ld, int lane) {
  int m = lane & 15, hi = lane >> 4;
  const __bf16* p = base + (size_t)m * ld + hi * 8;
  v8bf lo = *(const v8bf*)p;
  v8bf up = *(const v8bf*)(p + 16);
  return cat8(lo, up);
}

// B (32x16) sourced from B^T stored row-major [N][K]: lane n=lane&15 holds
// column n; hi = lane>>4 selects K groups {hi*8..hi*8+7} and {16+hi*8..16+hi*8+7}.
__device__ __forceinline__ v16bf load_tileB(const __bf16* baseT, int ld, int lane) {
  int nn = lane & 15, hi = lane >> 4;
  const __bf16* p = baseT + (size_t)nn * ld + hi * 8;
  v8bf lo = *(const v8bf*)p;
  v8bf up = *(const v8bf*)(p + 16);
  return cat8(lo, up);
}

// ---------------------------------------------------------------------------
// Generic GEMM: C[M][N] f32 = A(bf16 row-major) x B, with B given transposed
// (BT row-major [N][K], bf16). Each wave: 32 rows x 64 cols (2x4 acc tiles),
// so every B-tile load is reused by two WMMAs: 12 b128 loads per 8 WMMAs.
// Block 256 threads = 8 waves => 256x64 block tile.
// ---------------------------------------------------------------------------
__global__ void __launch_bounds__(256) gemm_bf16_wmma(
    const __bf16* __restrict__ A, const __bf16* __restrict__ BT,
    float* __restrict__ C, int M, int N, int K, int lda, int ldb, int ldc) {
  int lane = threadIdx.x & 31;
  int wave = threadIdx.x >> 5;
  int m0 = (blockIdx.y * 8 + wave) * 32;
  int n0 = blockIdx.x * 64;
  if (m0 >= M || n0 >= N) return;

  v8f acc[2][4] = {};
  for (int k0 = 0; k0 < K; k0 += 32) {
    v16bf a0 = load_tileA(A + (size_t)(m0 +  0) * lda + k0, lda, lane);
    v16bf a1 = load_tileA(A + (size_t)(m0 + 16) * lda + k0, lda, lane);
    #pragma unroll
    for (int t = 0; t < 4; ++t) {
      v16bf b = load_tileB(BT + (size_t)(n0 + t * 16) * ldb + k0, ldb, lane);
      acc[0][t] = __builtin_amdgcn_wmma_f32_16x16x32_bf16(false, a0, false, b,
                                                          (short)0, acc[0][t], false, false);
      acc[1][t] = __builtin_amdgcn_wmma_f32_16x16x32_bf16(false, a1, false, b,
                                                          (short)0, acc[1][t], false, false);
    }
  }

  int nn = lane & 15, hi = lane >> 4;
  #pragma unroll
  for (int mi = 0; mi < 2; ++mi) {
    #pragma unroll
    for (int j = 0; j < 8; ++j) {
      size_t row = (size_t)(m0 + mi * 16 + j + hi * 8) * ldc;
      #pragma unroll
      for (int t = 0; t < 4; ++t)
        C[row + n0 + t * 16 + nn] = acc[mi][t][j];
    }
  }
}

// ---------------------------------------------------------------------------
// Elementwise helpers
// ---------------------------------------------------------------------------
__global__ void f32_to_bf16(const float* __restrict__ in, __bf16* __restrict__ out,
                            size_t n) {
  size_t i = (size_t)blockIdx.x * blockDim.x + threadIdx.x;
  if (i < n) out[i] = (__bf16)in[i];
}

// in: [R][C] f32 row-major -> out: [C][R] bf16 row-major (i.e. out = in^T)
__global__ void transpose_to_bf16(const float* __restrict__ in, __bf16* __restrict__ out,
                                  int R, int Ccols) {
  size_t i = (size_t)blockIdx.x * blockDim.x + threadIdx.x;
  if (i >= (size_t)R * Ccols) return;
  int c = (int)(i / R);
  int r = (int)(i % R);
  out[(size_t)c * R + r] = (__bf16)in[(size_t)r * Ccols + c];
}

// GPT-NeoX half-rotation RoPE over q and k slices of qkv; writes bf16 q/k.
__global__ void rope_split(const float* __restrict__ qkv, const int* __restrict__ pos,
                           __bf16* __restrict__ qb, __bf16* __restrict__ kb, int total) {
  int i = blockIdx.x * blockDim.x + threadIdx.x;
  if (i >= total) return;                 // total = S*(NH+NKV)*64
  int d  = i & 63;
  int t  = i >> 6;
  int hh = t % (NH_ + NKV_);
  int s  = t / (NH_ + NKV_);
  const float* base = (hh < NH_)
      ? qkv + (size_t)s * QKVW + hh * D_
      : qkv + (size_t)s * QKVW + NH_ * D_ + (hh - NH_) * D_;
  float x1 = base[d], x2 = base[d + 64];
  float p = (float)pos[s];
  // inv_freq = theta^(-d/64) = exp(-ln(10000)/64 * d)
  float f = p * __expf(-0.14391156f * (float)d);
  float sn, cs;
  __sincosf(f, &sn, &cs);
  float r1 = x1 * cs - x2 * sn;
  float r2 = x2 * cs + x1 * sn;
  if (hh < NH_) {
    __bf16* o = qb + (size_t)s * (NH_ * D_) + hh * D_;
    o[d] = (__bf16)r1; o[d + 64] = (__bf16)r2;
  } else {
    __bf16* o = kb + (size_t)s * (NKV_ * D_) + (hh - NH_) * D_;
    o[d] = (__bf16)r1; o[d + 64] = (__bf16)r2;
  }
}

// V -> Vt[NKV][D][S] bf16 (so PV WMMA B-operand loads are contiguous over keys)
__global__ void v_transpose(const float* __restrict__ qkv, __bf16* __restrict__ vt,
                            int total) {
  int i = blockIdx.x * blockDim.x + threadIdx.x;
  if (i >= total) return;                 // total = NKV*D*S
  int s  = i % S_;
  int t  = i / S_;
  int d  = t % D_;
  int nv = t / D_;
  vt[(size_t)nv * D_ * S_ + (size_t)d * S_ + s] =
      (__bf16)qkv[(size_t)s * QKVW + (size_t)(NH_ + NKV_) * D_ + nv * D_ + d];
}

// ---------------------------------------------------------------------------
// Flash-style GQA sliding-window attention.
// One wave handles one (head, 16-query tile). 4 waves/block = 4 heads sharing
// the same q-tile (identical trip counts). Online softmax in f32; P staged
// through LDS to convert C-layout -> A-layout for the PV WMMAs.
// Q: [S][NH*D] bf16 (post-RoPE), K: [S][NKV*D] bf16, Vt: [NKV][D][S] bf16,
// O: [S][NH*D] bf16.
// ---------------------------------------------------------------------------
__global__ void __launch_bounds__(128) attn_wmma(
    const __bf16* __restrict__ Q, const __bf16* __restrict__ Kc,
    const __bf16* __restrict__ Vt, __bf16* __restrict__ O) {
  __shared__ __attribute__((aligned(16))) __bf16 Plds[4][16 * 32];
  int lane = threadIdx.x & 31;
  int wave = threadIdx.x >> 5;
  int nn = lane & 15, hi = lane >> 4;
  int q0 = blockIdx.x * 16;
  int h  = blockIdx.y * 4 + wave;
  int kvh = h >> 2;                               // GQA group size G = 4
  const __bf16* qbase = Q  + (size_t)h   * D_;    // ld = NH*D
  const __bf16* kbase = Kc + (size_t)kvh * D_;    // ld = NKV*D
  const __bf16* vbase = Vt + (size_t)kvh * D_ * S_;  // ld = S
  __bf16* myP = &Plds[wave][0];

  // Q tile: 16 x 128 as four 16x32 A operands, kept in registers.
  v16bf aq[4];
  #pragma unroll
  for (int kc = 0; kc < 4; ++kc)
    aq[kc] = load_tileA(qbase + (size_t)q0 * (NH_ * D_) + kc * 32, NH_ * D_, lane);

  v8f o[8] = {};                                  // 16x128 f32 accumulator
  float mrow[8], lrow[8];
  #pragma unroll
  for (int j = 0; j < 8; ++j) { mrow[j] = -3.0e38f; lrow[j] = 0.0f; }

  const float scale = 0.08838834764831845f;       // 1/sqrt(128)
  int jmin = q0 - WIN_ + 1; if (jmin < 0) jmin = 0;
  int c0   = jmin & ~31;                          // 32-key chunk aligned start
  int cend = q0 + 16;                             // last valid key is q0+15

  for (int t0 = c0; t0 < cend; t0 += 32) {
    float sm[2][8];
    #pragma unroll
    for (int tt = 0; tt < 2; ++tt) {
      int kt = t0 + tt * 16;
      v8f sc = {};
      #pragma unroll
      for (int kc = 0; kc < 4; ++kc) {
        v16bf bk = load_tileB(kbase + (size_t)kt * (NKV_ * D_) + kc * 32,
                              NKV_ * D_, lane);
        sc = __builtin_amdgcn_wmma_f32_16x16x32_bf16(false, aq[kc], false, bk,
                                                     (short)0, sc, false, false);
      }
      #pragma unroll
      for (int j = 0; j < 8; ++j) {
        int iq = q0 + j + hi * 8;
        int jk = kt + nn;
        bool valid = (jk <= iq) && ((iq - jk) < WIN_);
        sm[tt][j] = valid ? sc[j] * scale : -3.0e38f;
      }
    }
    // Online softmax update per row (row lives across 16 lanes of a half).
    #pragma unroll
    for (int j = 0; j < 8; ++j) {
      float rm = fmaxf(sm[0][j], sm[1][j]);
      #pragma unroll
      for (int off = 8; off > 0; off >>= 1)
        rm = fmaxf(rm, __shfl_xor(rm, off, 16));
      float mnew = fmaxf(mrow[j], rm);
      float corr = __expf(mrow[j] - mnew);
      float p0 = (sm[0][j] > -1.0e37f) ? __expf(sm[0][j] - mnew) : 0.0f;
      float p1 = (sm[1][j] > -1.0e37f) ? __expf(sm[1][j] - mnew) : 0.0f;
      float rs = p0 + p1;
      #pragma unroll
      for (int off = 8; off > 0; off >>= 1)
        rs += __shfl_xor(rs, off, 16);
      lrow[j] = lrow[j] * corr + rs;
      mrow[j] = mnew;
      #pragma unroll
      for (int dt = 0; dt < 8; ++dt) o[dt][j] *= corr;
      // Store P (C-layout -> row-major 16x32 in LDS) for A-operand reload.
      myP[(j + hi * 8) * 32 + nn]      = (__bf16)p0;
      myP[(j + hi * 8) * 32 + 16 + nn] = (__bf16)p1;
    }
    asm volatile("s_wait_dscnt 0" ::: "memory");  // cross-lane LDS RAW fence
    v16bf ap = load_tileA(myP, 32, lane);
    #pragma unroll
    for (int dt = 0; dt < 8; ++dt) {
      v16bf bv = load_tileB(vbase + (size_t)(dt * 16) * S_ + t0, S_, lane);
      o[dt] = __builtin_amdgcn_wmma_f32_16x16x32_bf16(false, ap, false, bv,
                                                      (short)0, o[dt], false, false);
    }
    asm volatile("s_wait_dscnt 0" ::: "memory");  // loads done before next overwrite
  }

  #pragma unroll
  for (int dt = 0; dt < 8; ++dt) {
    #pragma unroll
    for (int j = 0; j < 8; ++j) {
      float val = o[dt][j] / lrow[j];
      O[(size_t)(q0 + j + hi * 8) * (NH_ * D_) + h * D_ + dt * 16 + nn] = (__bf16)val;
    }
  }
}

// ---------------------------------------------------------------------------
// Host launcher
// ---------------------------------------------------------------------------
extern "C" void kernel_launch(void* const* d_in, const int* in_sizes, int n_in,
                              void* d_out, int out_size, void* d_ws, size_t ws_size,
                              hipStream_t stream) {
  const int*   positions = (const int*)d_in[0];
  const float* hidden    = (const float*)d_in[1];
  const float* w_qkv     = (const float*)d_in[2];
  const float* w_o       = (const float*)d_in[3];
  float* out = (float*)d_out;

  char* ws = (char*)d_ws;
  size_t off = 0;
  auto alloc = [&](size_t bytes) {
    void* p = ws + off;
    off += (bytes + 255) & ~(size_t)255;
    return p;
  };
  __bf16* hb    = (__bf16*)alloc((size_t)S_ * H_ * 2);        // hidden bf16
  __bf16* wqkvT = (__bf16*)alloc((size_t)QKVW * H_ * 2);      // w_qkv^T bf16
  __bf16* woT   = (__bf16*)alloc((size_t)H_ * H_ * 2);        // w_o^T bf16
  float*  qkv   = (float*) alloc((size_t)S_ * QKVW * 4);      // qkv f32
  __bf16* qb    = (__bf16*)alloc((size_t)S_ * NH_ * D_ * 2);  // q (rope) bf16
  __bf16* kb    = (__bf16*)alloc((size_t)S_ * NKV_ * D_ * 2); // k (rope) bf16
  __bf16* vt    = (__bf16*)alloc((size_t)NKV_ * D_ * S_ * 2); // v^T bf16
  __bf16* ab    = (__bf16*)alloc((size_t)S_ * NH_ * D_ * 2);  // attn out bf16

  // 1. precision conversion + weight transposes (one pass, L2-resident after)
  { size_t n = (size_t)S_ * H_;
    f32_to_bf16<<<(unsigned)((n + 255) / 256), 256, 0, stream>>>(hidden, hb, n); }
  { size_t n = (size_t)H_ * QKVW;
    transpose_to_bf16<<<(unsigned)((n + 255) / 256), 256, 0, stream>>>(w_qkv, wqkvT, H_, QKVW); }
  { size_t n = (size_t)H_ * H_;
    transpose_to_bf16<<<(unsigned)((n + 255) / 256), 256, 0, stream>>>(w_o, woT, H_, H_); }

  // 2. QKV projection: [2048,4096] x [4096,6144] -> f32
  gemm_bf16_wmma<<<dim3(QKVW / 64, S_ / 256), 256, 0, stream>>>(
      hb, wqkvT, qkv, S_, QKVW, H_, H_, H_, QKVW);

  // 3. RoPE on q/k + V transpose (f32 math, bf16 outputs)
  { int n = S_ * (NH_ + NKV_) * 64;
    rope_split<<<(n + 255) / 256, 256, 0, stream>>>(qkv, positions, qb, kb, n); }
  { int n = NKV_ * D_ * S_;
    v_transpose<<<(n + 255) / 256, 256, 0, stream>>>(qkv, vt, n); }

  // 4. sliding-window GQA flash attention
  attn_wmma<<<dim3(S_ / 16, NH_ / 4), 128, 0, stream>>>(qb, kb, vt, ab);

  // 5. output projection: [2048,4096] x [4096,4096] -> f32 d_out
  gemm_bf16_wmma<<<dim3(H_ / 64, S_ / 256), 256, 0, stream>>>(
      ab, woT, out, S_, H_, H_, H_, H_, H_);
}
